// MobileViTBlockv2_24747601560092
// MI455X (gfx1250) — matrix operand order repaired and checked by reading, not compiled, not run on hardware
//
#include <hip/hip_runtime.h>

typedef __attribute__((ext_vector_type(16))) _Float16 v16h;
typedef __attribute__((ext_vector_type(8)))  _Float16 v8h;
typedef __attribute__((ext_vector_type(8)))  float    v8f;

static constexpr int  BDIM = 32;
static constexpr int  CDIM = 128;
static constexpr int  HDIM = 64;
static constexpr int  WDIM = 64;
static constexpr int  DDIM = 256;
static constexpr int  FFD  = 512;
static constexpr long RTOT = (long)BDIM * HDIM * WDIM;   // 131072 token rows

static constexpr int BM = 128, BN = 128, BK = 32;

// ---------------------------------------------------------------------------
// CDNA5 async global->LDS copy (ASYNCcnt-tracked), guarded so the file still
// compiles if the toolchain lacks the builtin. The builtin takes typed
// int4-vector pointers: (v4i addrspace(1)*, v4i addrspace(3)*, imm, imm).
// ---------------------------------------------------------------------------
#if defined(__has_builtin)
#if __has_builtin(__builtin_amdgcn_global_load_async_to_lds_b128)
#define HAVE_ASYNC_LDS 1
#endif
#endif
#ifndef HAVE_ASYNC_LDS
#define HAVE_ASYNC_LDS 0
#endif

#if HAVE_ASYNC_LDS
#if __has_builtin(__builtin_amdgcn_s_wait_asynccnt)
#define WAIT_ASYNC(n) __builtin_amdgcn_s_wait_asynccnt(n)
#else
#define WAIT_ASYNC(n) asm volatile("s_wait_asynccnt %0" ::"i"(n) : "memory")
#endif
typedef int async_v4i __attribute__((vector_size(16)));
__device__ __forceinline__ void async_ld16(const void* gsrc, void* ldst) {
  __builtin_amdgcn_global_load_async_to_lds_b128(
      (__attribute__((address_space(1))) async_v4i*)(unsigned long long)gsrc,
      (__attribute__((address_space(3))) async_v4i*)(unsigned int)(unsigned long long)ldst,
      0, 0);
}
#endif

// ---------------------------------------------------------------------------
// f32 -> f16 weight conversion
// ---------------------------------------------------------------------------
__global__ void k_cvt(const float* __restrict__ s, _Float16* __restrict__ d, int n) {
  int i = blockIdx.x * 256 + threadIdx.x;
  if (i < n) d[i] = (_Float16)s[i];
}

__global__ void k_zero(float* __restrict__ p) {
  p[blockIdx.x * 256 + threadIdx.x] = 0.f;
}

// ---------------------------------------------------------------------------
// Depthwise 3x3 + BN + leakyReLU, output f16 in patch-token row order [r, c].
// Lanes sweep w (coalesced f32 reads); LDS-staged transpose so the f16 matrix
// is written in coalesced 16B chunks.
// grid = (B*H, C/32), block = 256 (64 w  x  4 channel-quads)
// ---------------------------------------------------------------------------
__global__ __launch_bounds__(256) void k_dw(
    const float* __restrict__ x, const float* __restrict__ w,
    const float* __restrict__ g, const float* __restrict__ bb,
    const float* __restrict__ m, const float* __restrict__ v,
    _Float16* __restrict__ Xp)
{
  __shared__ _Float16 sh[64][40];               // [w][c-in-group], 16B-aligned rows
  const int bh = blockIdx.x;
  const int b  = bh >> 6, h = bh & 63;
  const int cg = blockIdx.y * 32;
  const int wI = threadIdx.x & 63;
  const int cq = threadIdx.x >> 6;              // 0..3

#pragma unroll
  for (int i = 0; i < 8; ++i) {
    const int c = cg + cq * 8 + i;
    const float* xb = x + ((long)(b * CDIM + c)) * (HDIM * WDIM);
    float acc = 0.f;
#pragma unroll
    for (int kh = 0; kh < 3; ++kh) {
      const int hh = h + kh - 1;
      if ((unsigned)hh < (unsigned)HDIM) {
        const float* row = xb + hh * WDIM;
#pragma unroll
        for (int kw = 0; kw < 3; ++kw) {
          const int wq = wI + kw - 1;
          if ((unsigned)wq < (unsigned)WDIM)
            acc += row[wq] * w[c * 9 + kh * 3 + kw];
        }
      }
    }
    float y = (acc - m[c]) * rsqrtf(v[c] + 1e-5f) * g[c] + bb[c];
    y = y > 0.f ? y : 0.1f * y;
    sh[wI][cq * 8 + i] = (_Float16)y;
  }
  __syncthreads();

  const int row_w = threadIdx.x >> 2;           // w of the token row we emit
  const int part  = threadIdx.x & 3;            // 8-channel chunk
  const int s = (h & 1) * 2 + (row_w & 1);
  const int n = (h >> 1) * 32 + (row_w >> 1);
  const long rp = ((long)b * 4 + s) * 1024 + n;
  v8h o;
#pragma unroll
  for (int i = 0; i < 8; ++i) o[i] = sh[row_w][part * 8 + i];
  *(v8h*)&Xp[rp * CDIM + cg + part * 8] = o;
}

// ---------------------------------------------------------------------------
// WMMA GEMM: Y[R,M] = X[R,K] * Wt[M,K]^T, f16 in, f32 accumulate.
// Double-buffered LDS tiles; async global->LDS staging when available.
// EPI 0: +bias, store f16       EPI 1: +bias + residual (Y read+write f16)
// EPI 2: +bias, leakyReLU(0.1)  EPI 3: BN + fold to NCHW f32 (proj)
// ---------------------------------------------------------------------------
template<int EPI>
__global__ __launch_bounds__(256) void k_gemm(
    const _Float16* __restrict__ X, const _Float16* __restrict__ Wt,
    const float* __restrict__ bias, _Float16* __restrict__ Y,
    float* __restrict__ outf,
    const float* __restrict__ bng, const float* __restrict__ bnb,
    const float* __restrict__ bnm, const float* __restrict__ bnvv,
    int K, int M)
{
  __shared__ _Float16 Xs[2][BM][BK + 8];
  __shared__ _Float16 Ws[2][BN][BK + 8];

  const int tid  = threadIdx.x;
  const int lane = tid & 31;
  const int wid  = tid >> 5;
  const int wr   = wid >> 1;          // 0..3  -> 32-row strip
  const int wc   = wid & 1;           // 0..1  -> 64-col strip
  const long rb  = (long)blockIdx.x * BM;
  const long cb  = (long)blockIdx.y * BN;

  v8f acc[2][4];
#pragma unroll
  for (int i = 0; i < 2; ++i)
#pragma unroll
    for (int j = 0; j < 4; ++j)
      acc[i][j] = (v8f){0.f,0.f,0.f,0.f,0.f,0.f,0.f,0.f};

  const int rowl = tid >> 1;          // 0..127
  const int c16  = (tid & 1) * 16;    // 0 or 16 (f16 elems)

  auto stage = [&](int buf, int kt) {
    const _Float16* xg = X  + (rb + rowl) * (long)K + kt + c16;
    const _Float16* wg = Wt + (cb + rowl) * (long)K + kt + c16;
#if HAVE_ASYNC_LDS
    async_ld16(xg,     &Xs[buf][rowl][c16]);
    async_ld16(xg + 8, &Xs[buf][rowl][c16 + 8]);
    async_ld16(wg,     &Ws[buf][rowl][c16]);
    async_ld16(wg + 8, &Ws[buf][rowl][c16 + 8]);
#else
    float4 x0 = ((const float4*)xg)[0], x1 = ((const float4*)xg)[1];
    float4 w0 = ((const float4*)wg)[0], w1 = ((const float4*)wg)[1];
    *(float4*)&Xs[buf][rowl][c16]     = x0;
    *(float4*)&Xs[buf][rowl][c16 + 8] = x1;
    *(float4*)&Ws[buf][rowl][c16]     = w0;
    *(float4*)&Ws[buf][rowl][c16 + 8] = w1;
#endif
  };

  const int nk = K / BK;
  stage(0, 0);

  for (int it = 0; it < nk; ++it) {
    const int cur = it & 1;
    if (it + 1 < nk) {
      stage(cur ^ 1, (it + 1) * BK);        // keep next slab in flight
#if HAVE_ASYNC_LDS
      WAIT_ASYNC(4);                        // 4 newest may still be pending
#endif
    } else {
#if HAVE_ASYNC_LDS
      WAIT_ASYNC(0);
#endif
    }
    __syncthreads();                        // current tiles visible to all waves

    const int mfrag = lane & 15;
    const int kbA = (lane >> 4) * 8;   // A: lanes 0-15 K[0..7]&[16..23], lanes 16-31 +8
    const int kbB = (lane >> 4) * 8;   // B: lanes 0-15 K[0..7]&[16..23], lanes 16-31 +8

    v16h afr[2], bfr[4];
#pragma unroll
    for (int rt = 0; rt < 2; ++rt) {
      const _Float16* p = &Xs[cur][wr * 32 + rt * 16 + mfrag][0];
      v8h lo = *(const v8h*)(p + kbA);
      v8h hi = *(const v8h*)(p + kbA + 16);
#pragma unroll
      for (int i = 0; i < 8; ++i) { afr[rt][i] = lo[i]; afr[rt][i + 8] = hi[i]; }
    }
#pragma unroll
    for (int ct = 0; ct < 4; ++ct) {
      const _Float16* p = &Ws[cur][wc * 64 + ct * 16 + mfrag][0];
      v8h lo = *(const v8h*)(p + kbB);
      v8h hi = *(const v8h*)(p + kbB + 16);
#pragma unroll
      for (int i = 0; i < 8; ++i) { bfr[ct][i] = lo[i]; bfr[ct][i + 8] = hi[i]; }
    }
#pragma unroll
    for (int rt = 0; rt < 2; ++rt)
#pragma unroll
      for (int ct = 0; ct < 4; ++ct)
        acc[rt][ct] = __builtin_amdgcn_wmma_f32_16x16x32_f16(
            false, afr[rt], false, bfr[ct], (short)0, acc[rt][ct], false, false);

    __syncthreads();                        // all waves done before buffer reuse
  }

  // epilogue: C/D layout -> lane holds N=lane&15, VGPR j -> M = j + 8*(lane>=16)
  const int laneN = lane & 15;
  const int hi8   = (lane >> 4) * 8;
#pragma unroll
  for (int rt = 0; rt < 2; ++rt) {
#pragma unroll
    for (int ct = 0; ct < 4; ++ct) {
#pragma unroll
      for (int j = 0; j < 8; ++j) {
        long row = rb + wr * 32 + rt * 16 + j + hi8;
        long col = cb + wc * 64 + ct * 16 + laneN;
        float v = acc[rt][ct][j];
        if (EPI != 3) v += bias[col];
        if (EPI == 0) {
          Y[row * (long)M + col] = (_Float16)v;
        } else if (EPI == 1) {
          long idx = row * (long)M + col;
          Y[idx] = (_Float16)((float)Y[idx] + v);
        } else if (EPI == 2) {
          v = v > 0.f ? v : 0.1f * v;
          Y[row * (long)M + col] = (_Float16)v;
        } else {
          float o = (v - bnm[col]) * rsqrtf(bnvv[col] + 1e-5f) * bng[col] + bnb[col];
          long b = row >> 12, s = (row >> 10) & 3, n = row & 1023;
          long h = ((n >> 5) << 1) + (s >> 1);
          long w = ((n & 31) << 1) + (s & 1);
          outf[((b * CDIM + col) << 12) | (h << 6) | w] = o;   // absorbed by 192MB L2
        }
      }
    }
  }
}

// ---------------------------------------------------------------------------
// Channel-first LayerNorm over D=256, one wave per row
// ---------------------------------------------------------------------------
__global__ __launch_bounds__(256) void k_ln(
    const _Float16* __restrict__ X, _Float16* __restrict__ Y,
    const float* __restrict__ g, const float* __restrict__ b)
{
  long row = (long)blockIdx.x * 8 + (threadIdx.x >> 5);
  int lane = threadIdx.x & 31;
  v8h xv = *(const v8h*)(X + row * DDIM + lane * 8);
  float f[8], s = 0.f, ss = 0.f;
#pragma unroll
  for (int i = 0; i < 8; ++i) { f[i] = (float)xv[i]; s += f[i]; ss += f[i] * f[i]; }
#pragma unroll
  for (int off = 16; off > 0; off >>= 1) { s += __shfl_xor(s, off); ss += __shfl_xor(ss, off); }
  float mean = s * (1.f / DDIM);
  float var  = ss * (1.f / DDIM) - mean * mean;
  float inv  = rsqrtf(var + 1e-5f);
  v8h o;
#pragma unroll
  for (int i = 0; i < 8; ++i) {
    int d = lane * 8 + i;
    o[i] = (_Float16)((f[i] - mean) * inv * g[d] + b[d]);
  }
  *(v8h*)(Y + row * DDIM + lane * 8) = o;
}

// ---------------------------------------------------------------------------
// q[r] = Z[r,:] . wq + bq   (the 1-wide slice of the 513-wide qkv projection)
// ---------------------------------------------------------------------------
__global__ __launch_bounds__(256) void k_qdot(
    const _Float16* __restrict__ Z, const float* __restrict__ wq,
    const float* __restrict__ bq, float* __restrict__ q)
{
  long row = (long)blockIdx.x * 8 + (threadIdx.x >> 5);
  int lane = threadIdx.x & 31;
  v8h zv = *(const v8h*)(Z + row * DDIM + lane * 8);
  float s = 0.f;
#pragma unroll
  for (int i = 0; i < 8; ++i) s += (float)zv[i] * wq[lane * 8 + i];
#pragma unroll
  for (int off = 16; off > 0; off >>= 1) s += __shfl_xor(s, off);
  if (lane == 0) q[row] = s + bq[0];
}

// ---------------------------------------------------------------------------
// softmax over each group of 1024 tokens (in place)
// ---------------------------------------------------------------------------
__global__ __launch_bounds__(256) void k_softmax(float* __restrict__ q)
{
  __shared__ float sh[8];
  long base = (long)blockIdx.x * 1024;
  int t = threadIdx.x;
  float v[4], mx = -3.4e38f;
#pragma unroll
  for (int i = 0; i < 4; ++i) { v[i] = q[base + t + i * 256]; mx = fmaxf(mx, v[i]); }
#pragma unroll
  for (int off = 16; off > 0; off >>= 1) mx = fmaxf(mx, __shfl_xor(mx, off));
  if ((t & 31) == 0) sh[t >> 5] = mx;
  __syncthreads();
  mx = sh[0];
#pragma unroll
  for (int i = 1; i < 8; ++i) mx = fmaxf(mx, sh[i]);
  __syncthreads();
  float sum = 0.f;
#pragma unroll
  for (int i = 0; i < 4; ++i) { v[i] = __expf(v[i] - mx); sum += v[i]; }
#pragma unroll
  for (int off = 16; off > 0; off >>= 1) sum += __shfl_xor(sum, off);
  if ((t & 31) == 0) sh[t >> 5] = sum;
  __syncthreads();
  sum = 0.f;
#pragma unroll
  for (int i = 0; i < 8; ++i) sum += sh[i];
  float inv = 1.f / sum;
#pragma unroll
  for (int i = 0; i < 4; ++i) q[base + t + i * 256] = v[i] * inv;
}

// ---------------------------------------------------------------------------
// ctx[g,d] += sum_n K[g*1024+n, d] * score[g*1024+n]   (token axis split x8)
// ---------------------------------------------------------------------------
__global__ __launch_bounds__(256) void k_ctx(
    const _Float16* __restrict__ KV, const float* __restrict__ sc,
    float* __restrict__ ctx)
{
  const int g = blockIdx.x, d = threadIdx.x;
  const int n0 = blockIdx.y * 128;
  const _Float16* kp = KV + ((long)g * 1024 + n0) * FFD + d;
  const float* sp = sc + (long)g * 1024 + n0;
  float a = 0.f;
#pragma unroll 4
  for (int n = 0; n < 128; ++n) a += (float)kp[(long)n * FFD] * sp[n];
  atomicAdd(&ctx[g * DDIM + d], a);
}

// ---------------------------------------------------------------------------
// A_in[r,d] = relu(V[r,d]) * ctx[g(r),d]    (V = cols 256..511 of KV)
// ---------------------------------------------------------------------------
__global__ __launch_bounds__(256) void k_avin(
    const _Float16* __restrict__ KV, const float* __restrict__ ctx,
    _Float16* __restrict__ A)
{
  long i = ((long)blockIdx.x * 256 + threadIdx.x) * 8;
  long r = i >> 8;
  int d = (int)(i & 255);
  int g = (int)(r >> 10);
  v8h vv = *(const v8h*)&KV[r * FFD + DDIM + d];
  v8h o;
#pragma unroll
  for (int j = 0; j < 8; ++j) {
    float f = (float)vv[j];
    f = f > 0.f ? f : 0.f;
    o[j] = (_Float16)(f * ctx[g * DDIM + d + j]);
  }
  *(v8h*)&A[r * DDIM + d] = o;
}

// ---------------------------------------------------------------------------
extern "C" void kernel_launch(void* const* d_in, const int* in_sizes, int n_in,
                              void* d_out, int out_size, void* d_ws, size_t ws_size,
                              hipStream_t stream)
{
  const float* x     = (const float*)d_in[0];
  const float* dw_w  = (const float*)d_in[1];
  const float* bn1g  = (const float*)d_in[2];
  const float* bn1b  = (const float*)d_in[3];
  const float* bn1m  = (const float*)d_in[4];
  const float* bn1v  = (const float*)d_in[5];
  const float* c1_w  = (const float*)d_in[6];
  const float* c1_b  = (const float*)d_in[7];
  const float* ln1g  = (const float*)d_in[8];
  const float* ln1b  = (const float*)d_in[9];
  const float* qkvw  = (const float*)d_in[10];
  const float* qkvb  = (const float*)d_in[11];
  const float* outw  = (const float*)d_in[12];
  const float* outb  = (const float*)d_in[13];
  const float* ln2g  = (const float*)d_in[14];
  const float* ln2b  = (const float*)d_in[15];
  const float* f1w   = (const float*)d_in[16];
  const float* f1b   = (const float*)d_in[17];
  const float* f2w   = (const float*)d_in[18];
  const float* f2b   = (const float*)d_in[19];
  const float* lnfg  = (const float*)d_in[20];
  const float* lnfb  = (const float*)d_in[21];
  const float* projw = (const float*)d_in[22];
  const float* bn2g  = (const float*)d_in[23];
  const float* bn2b  = (const float*)d_in[24];
  const float* bn2m  = (const float*)d_in[25];
  const float* bn2v  = (const float*)d_in[26];

  const long R = RTOT;
  char* ws = (char*)d_ws;
  _Float16* P    = (_Float16*)ws;            ws += (size_t)R * DDIM * 2;   // 67 MB
  _Float16* bufA = (_Float16*)ws;            ws += (size_t)R * FFD * 2;    // 134 MB (Xp / KV / Hf)
  _Float16* bufB = (_Float16*)ws;            ws += (size_t)R * DDIM * 2;   // 67 MB (Z / A_in)
  float*    qbuf = (float*)ws;               ws += (size_t)R * 4;
  float*    ctx  = (float*)ws;               ws += (size_t)128 * DDIM * 4;
  _Float16* wh   = (_Float16*)ws;

  _Float16* wh_c1 = wh;                    wh += DDIM * CDIM;
  _Float16* wh_kv[2], *wh_out[2], *wh_f1[2], *wh_f2[2];
  for (int l = 0; l < 2; ++l) {
    wh_kv[l]  = wh; wh += FFD * DDIM;
    wh_out[l] = wh; wh += DDIM * DDIM;
    wh_f1[l]  = wh; wh += FFD * DDIM;
    wh_f2[l]  = wh; wh += DDIM * FFD;
  }
  _Float16* wh_proj = wh;                  wh += CDIM * DDIM;

  auto cvt = [&](const float* s, _Float16* d, int n) {
    k_cvt<<<dim3((n + 255) / 256), dim3(256), 0, stream>>>(s, d, n);
  };
  cvt(c1_w, wh_c1, DDIM * CDIM);
  for (int l = 0; l < 2; ++l) {
    cvt(qkvw + (long)l * 513 * DDIM + DDIM, wh_kv[l], FFD * DDIM);  // rows 1..512
    cvt(outw + (long)l * DDIM * DDIM, wh_out[l], DDIM * DDIM);
    cvt(f1w  + (long)l * FFD * DDIM,  wh_f1[l],  FFD * DDIM);
    cvt(f2w  + (long)l * DDIM * FFD,  wh_f2[l],  DDIM * FFD);
  }
  cvt(projw, wh_proj, CDIM * DDIM);

  // depthwise + BN + leaky, output patch-ordered f16 [R, C] (lives in bufA)
  _Float16* Xp = bufA;
  k_dw<<<dim3(BDIM * HDIM, CDIM / 32), dim3(256), 0, stream>>>(
      x, dw_w, bn1g, bn1b, bn1m, bn1v, Xp);

  dim3 th(256);
  // conv1: P = Xp * c1_w^T + c1_b
  k_gemm<0><<<dim3(R / BM, DDIM / BN), th, 0, stream>>>(
      Xp, wh_c1, c1_b, P, nullptr, nullptr, nullptr, nullptr, nullptr, CDIM, DDIM);

  for (int l = 0; l < 2; ++l) {
    _Float16* Z  = bufB;
    _Float16* KV = bufA;
    k_ln<<<dim3(R / 8), th, 0, stream>>>(P, Z, ln1g + l * DDIM, ln1b + l * DDIM);
    k_qdot<<<dim3(R / 8), th, 0, stream>>>(Z, qkvw + (long)l * 513 * DDIM,
                                           qkvb + (long)l * 513, qbuf);
    k_gemm<0><<<dim3(R / BM, FFD / BN), th, 0, stream>>>(
        Z, wh_kv[l], qkvb + (long)l * 513 + 1, KV,
        nullptr, nullptr, nullptr, nullptr, nullptr, DDIM, FFD);
    k_softmax<<<dim3(128), th, 0, stream>>>(qbuf);
    k_zero<<<dim3(128), th, 0, stream>>>(ctx);
    k_ctx<<<dim3(128, 8), th, 0, stream>>>(KV, qbuf, ctx);
    k_avin<<<dim3(R / 8), th, 0, stream>>>(KV, ctx, Z);     // Z reused as A_in
    k_gemm<1><<<dim3(R / BM, DDIM / BN), th, 0, stream>>>(
        Z, wh_out[l], outb + l * DDIM, P,
        nullptr, nullptr, nullptr, nullptr, nullptr, DDIM, DDIM);
    k_ln<<<dim3(R / 8), th, 0, stream>>>(P, Z, ln2g + l * DDIM, ln2b + l * DDIM);
    _Float16* Hf = bufA;                                    // KV dead, reuse
    k_gemm<2><<<dim3(R / BM, FFD / BN), th, 0, stream>>>(
        Z, wh_f1[l], f1b + l * FFD, Hf,
        nullptr, nullptr, nullptr, nullptr, nullptr, DDIM, FFD);
    k_gemm<1><<<dim3(R / BM, DDIM / BN), th, 0, stream>>>(
        Hf, wh_f2[l], f2b + l * DDIM, P,
        nullptr, nullptr, nullptr, nullptr, nullptr, FFD, DDIM);
  }

  // final LN + proj + BN + fold back to NCHW f32
  k_ln<<<dim3(R / 8), th, 0, stream>>>(P, bufB, lnfg, lnfb);
  k_gemm<3><<<dim3(R / BM, CDIM / BN), th, 0, stream>>>(
      bufB, wh_proj, nullptr, nullptr, (float*)d_out,
      bn2g, bn2b, bn2m, bn2v, DDIM, CDIM);
}
